// PHMultiHeadAttention_51668456571098
// MI455X (gfx1250) — compile-verified, matrix operand order checked by reading, not run on hardware
//
#include <hip/hip_runtime.h>
#include <math.h>

typedef __attribute__((ext_vector_type(16))) __bf16 v16bf;
typedef __attribute__((ext_vector_type(8)))  __bf16 v8bf;
typedef __attribute__((ext_vector_type(8)))  float  v8f;
typedef __attribute__((ext_vector_type(4)))  float  v4f;

#define BN 4
#define HN 16
#define SN 1024
#define DM 1024
#define DK 64

static __device__ __forceinline__ v8f wmma_bf16(v16bf a, v16bf b, v8f c) {
    return __builtin_amdgcn_wmma_f32_16x16x32_bf16(false, a, false, b, (short)0, c,
                                                   false, false);
}

// ---------------------------------------------------------------------------
// Kernel 1: convert Wq/Wk (f32 row-major [k][n]) -> bf16 transposed WT[n][k]
// so the GEMM B-operand fragment loads are contiguous per lane.
// ---------------------------------------------------------------------------
__global__ void transpose_wb(const float* __restrict__ Wq,
                             const float* __restrict__ Wk,
                             __bf16* __restrict__ WqT,
                             __bf16* __restrict__ WkT) {
    int n = blockIdx.x * 32 + threadIdx.x;
    int k = blockIdx.y * 8 + threadIdx.y;
    const float* W = blockIdx.z ? Wk : Wq;
    __bf16* WT = blockIdx.z ? WkT : WqT;
    WT[(size_t)n * DM + k] = (__bf16)W[(size_t)k * DM + n];
}

// ---------------------------------------------------------------------------
// Kernel 2: a[b][d] = aspect[b] . Wd[:,d] + bd[d]   (tiny: 256 outputs)
// ---------------------------------------------------------------------------
__global__ void aspect_proj(const float* __restrict__ aspect,
                            const float* __restrict__ Wd,
                            const float* __restrict__ bd,
                            float* __restrict__ aw) {
    int tid = threadIdx.x;          // 256 threads
    int b = tid >> 6, d = tid & 63;
    float s = bd[d];
    for (int k = 0; k < DM; ++k)
        s += aspect[(size_t)b * DM + k] * Wd[(size_t)k * DK + d];
    aw[tid] = s;
}

// ---------------------------------------------------------------------------
// Kernel 3: projection GEMM.  One wave computes a 16x64 strip of
// Out[4096,1024]bf16 = A[4096,1024]f32 @ W + bias, via v_wmma_f32_16x16x32_bf16.
// blockIdx.z selects (query,Wq,bq,Qbf) vs (key,Wk,bk,Kbf).
// ---------------------------------------------------------------------------
__global__ __launch_bounds__(32) void proj_gemm(
    const float* __restrict__ qin, const float* __restrict__ kin,
    const __bf16* __restrict__ WqT, const __bf16* __restrict__ WkT,
    const float* __restrict__ bq, const float* __restrict__ bk,
    __bf16* __restrict__ Qbf, __bf16* __restrict__ Kbf) {

    const int lane = threadIdx.x & 31;
    const int half = lane >> 4;
    const int l16  = lane & 15;
    const int m0 = blockIdx.x * 16;
    const int n0 = blockIdx.y * 64;

    const float*  A    = blockIdx.z ? kin : qin;
    const __bf16* WT   = blockIdx.z ? WkT : WqT;
    const float*  bias = blockIdx.z ? bk  : bq;
    __bf16*       Out  = blockIdx.z ? Kbf : Qbf;

    v8f acc[4] = {};

    const float* arow = A + (size_t)(m0 + l16) * DM;

    for (int kk = 0; kk < DM; kk += 32) {
        const int kbase = kk + half * 8;
        // A fragment: lanes 0-15 hold K = kk+{0..7,16..23}, lanes 16-31 +8.
        v4f c0 = *(const v4f*)(arow + kbase);
        v4f c1 = *(const v4f*)(arow + kbase + 4);
        v4f c2 = *(const v4f*)(arow + kbase + 16);
        v4f c3 = *(const v4f*)(arow + kbase + 20);
        v16bf af;
#pragma unroll
        for (int e = 0; e < 4; ++e) {
            af[e]      = (__bf16)c0[e];
            af[e + 4]  = (__bf16)c1[e];
            af[e + 8]  = (__bf16)c2[e];
            af[e + 12] = (__bf16)c3[e];
        }
#pragma unroll
        for (int j = 0; j < 4; ++j) {
            const __bf16* wr = WT + (size_t)(n0 + j * 16 + l16) * DM + kbase;
            v8bf b0 = *(const v8bf*)(wr);
            v8bf b1 = *(const v8bf*)(wr + 16);
            v16bf bv;
#pragma unroll
            for (int e = 0; e < 8; ++e) { bv[e] = b0[e]; bv[e + 8] = b1[e]; }
            acc[j] = wmma_bf16(af, bv, acc[j]);
        }
    }

#pragma unroll
    for (int j = 0; j < 4; ++j) {
        const int n = n0 + j * 16 + l16;
        const float bs = bias[n];
#pragma unroll
        for (int r = 0; r < 8; ++r) {
            const int row = m0 + r + half * 8;
            Out[(size_t)row * DM + n] = (__bf16)(acc[j][r] + bs);
        }
    }
}

// ---------------------------------------------------------------------------
// Kernel 4: asp[b][h][t] = tanh( a[b] . k[b,h,t,:] )
// ---------------------------------------------------------------------------
__global__ void aspect_scores(const float* __restrict__ aw,
                              const __bf16* __restrict__ Kbf,
                              float* __restrict__ asp) {
    int idx = blockIdx.x * 256 + threadIdx.x;   // 65536 total
    int t = idx & (SN - 1);
    int h = (idx >> 10) & (HN - 1);
    int b = idx >> 14;
    const __bf16* kr = Kbf + ((size_t)(b * SN + t)) * DM + h * DK;
    const float*  a  = aw + b * DK;
    float s = 0.f;
#pragma unroll 8
    for (int d = 0; d < DK; ++d) s += a[d] * (float)kr[d];
    asp[idx] = tanhf(s);
}

// ---------------------------------------------------------------------------
// Kernel 5: scores + aspect + softmax.  One block (8 waves) per (b,h,16 rows).
// Wave w owns column tiles t0 = (w*8+j)*16, j=0..7 -> full 1024 keys covered.
// ---------------------------------------------------------------------------
__global__ __launch_bounds__(256) void attn(const __bf16* __restrict__ Qbf,
                                            const __bf16* __restrict__ Kbf,
                                            const float* __restrict__ asp,
                                            float* __restrict__ out) {
    __shared__ float redmax[128];   // [row 0..15][wave 0..7]
    __shared__ float redsum[128];

    const int tid  = threadIdx.x;
    const int w    = tid >> 5;
    const int lane = tid & 31;
    const int half = lane >> 4;
    const int l16  = lane & 15;

    const int wg  = blockIdx.x;           // 0..4095
    const int s0  = (wg & 63) * 16;
    const int bh  = wg >> 6;
    const int h   = bh & (HN - 1);
    const int b   = bh >> 4;

    // ---- Q fragments (A operand), shared by all tiles of this wave ----
    const __bf16* qrow = Qbf + ((size_t)(b * SN + s0 + l16)) * DM + h * DK + half * 8;
    v16bf aq[2];
#pragma unroll
    for (int ks = 0; ks < 2; ++ks) {
        v8bf c0 = *(const v8bf*)(qrow + ks * 32);
        v8bf c1 = *(const v8bf*)(qrow + ks * 32 + 16);
#pragma unroll
        for (int e = 0; e < 8; ++e) { aq[ks][e] = c0[e]; aq[ks][e + 8] = c1[e]; }
    }

    const __bf16* kbase = Kbf + ((size_t)b * SN) * DM + h * DK + half * 8;
    const float*  aspbh = asp + ((size_t)(b * HN + h)) * SN;

    v8f acc[8];
#pragma unroll
    for (int j = 0; j < 8; ++j) {
        const int t0 = (w * 8 + j) * 16;
        const __bf16* krow = kbase + (size_t)(t0 + l16) * DM;
        v16bf bk0, bk1;
        v8bf p0 = *(const v8bf*)(krow);
        v8bf p1 = *(const v8bf*)(krow + 16);
        v8bf p2 = *(const v8bf*)(krow + 32);
        v8bf p3 = *(const v8bf*)(krow + 48);
#pragma unroll
        for (int e = 0; e < 8; ++e) {
            bk0[e] = p0[e]; bk0[e + 8] = p1[e];
            bk1[e] = p2[e]; bk1[e + 8] = p3[e];
        }
        v8f c = {};
        c = wmma_bf16(aq[0], bk0, c);
        c = wmma_bf16(aq[1], bk1, c);
        const float av = aspbh[t0 + l16];   // aspect term: depends on key t only
#pragma unroll
        for (int r = 0; r < 8; ++r) acc[j][r] = c[r] * 0.125f + av;  // 1/sqrt(64)
    }

    // ---- row-wise max: lanes within 16-lane half, then across 8 waves ----
    float rm[8];
#pragma unroll
    for (int r = 0; r < 8; ++r) {
        float v = acc[0][r];
#pragma unroll
        for (int j = 1; j < 8; ++j) v = fmaxf(v, acc[j][r]);
#pragma unroll
        for (int m = 1; m < 16; m <<= 1) v = fmaxf(v, __shfl_xor(v, m, 32));
        if (l16 == 0) redmax[(r + half * 8) * 8 + w] = v;
    }
    __syncthreads();
#pragma unroll
    for (int r = 0; r < 8; ++r) {
        const int row = r + half * 8;
        float v = redmax[row * 8 + 0];
#pragma unroll
        for (int q = 1; q < 8; ++q) v = fmaxf(v, redmax[row * 8 + q]);
        rm[r] = v;
    }

    // ---- exp + row-wise sum ----
    float inv[8];
#pragma unroll
    for (int r = 0; r < 8; ++r) {
        float s = 0.f;
#pragma unroll
        for (int j = 0; j < 8; ++j) {
            float e = __expf(acc[j][r] - rm[r]);
            acc[j][r] = e;
            s += e;
        }
#pragma unroll
        for (int m = 1; m < 16; m <<= 1) s += __shfl_xor(s, m, 32);
        if (l16 == 0) redsum[(r + half * 8) * 8 + w] = s;
    }
    __syncthreads();
#pragma unroll
    for (int r = 0; r < 8; ++r) {
        const int row = r + half * 8;
        float s = 0.f;
#pragma unroll
        for (int q = 1; q < 9; ++q) s += redsum[row * 8 + (q & 7)];
        inv[r] = 1.0f / s;
    }

    // ---- normalized store: out[b][h][s0+row][t] ----
    const size_t obase = ((size_t)(b * HN + h)) * SN * SN;
#pragma unroll
    for (int j = 0; j < 8; ++j) {
        const int t = (w * 8 + j) * 16 + l16;
#pragma unroll
        for (int r = 0; r < 8; ++r) {
            const int row = s0 + r + half * 8;
            out[obase + (size_t)row * SN + t] = acc[j][r] * inv[r];
        }
    }
}

// ---------------------------------------------------------------------------
extern "C" void kernel_launch(void* const* d_in, const int* in_sizes, int n_in,
                              void* d_out, int out_size, void* d_ws, size_t ws_size,
                              hipStream_t stream) {
    (void)in_sizes; (void)n_in; (void)out_size; (void)ws_size;
    const float* query  = (const float*)d_in[0];
    const float* key    = (const float*)d_in[1];
    const float* aspect = (const float*)d_in[2];
    const float* Wq     = (const float*)d_in[3];
    const float* bq     = (const float*)d_in[4];
    const float* Wk     = (const float*)d_in[5];
    const float* bk     = (const float*)d_in[6];
    const float* Wd     = (const float*)d_in[7];
    const float* bd     = (const float*)d_in[8];
    float* out = (float*)d_out;

    char* ws = (char*)d_ws;
    __bf16* Qbf = (__bf16*)(ws);                       // 8 MiB  [4096][1024]
    __bf16* Kbf = (__bf16*)(ws + (size_t)(8u  << 20)); // 8 MiB
    __bf16* WqT = (__bf16*)(ws + (size_t)(16u << 20)); // 2 MiB  [n][k]
    __bf16* WkT = (__bf16*)(ws + (size_t)(18u << 20)); // 2 MiB
    float*  aw  = (float*)(ws + (size_t)(20u << 20));  // 1 KiB  [B][DK]
    float*  asp = (float*)(ws + (size_t)(20u << 20) + 4096); // 256 KiB [B][H][S]

    transpose_wb<<<dim3(DM / 32, DM / 8, 2), dim3(32, 8), 0, stream>>>(Wq, Wk, WqT, WkT);
    aspect_proj<<<1, 256, 0, stream>>>(aspect, Wd, bd, aw);
    proj_gemm<<<dim3((BN * SN) / 16, DM / 64, 2), 32, 0, stream>>>(
        query, key, WqT, WkT, bq, bk, Qbf, Kbf);
    aspect_scores<<<(BN * HN * SN) / 256, 256, 0, stream>>>(aw, Kbf, asp);
    attn<<<BN * HN * (SN / 16), 256, 0, stream>>>(Qbf, Kbf, asp, out);
}
